// MultiHeadAttention_81793357185469
// MI455X (gfx1250) — matrix operand order repaired and checked by reading, not compiled, not run on hardware
//
#include <hip/hip_runtime.h>

typedef __attribute__((ext_vector_type(8)))  float        v8f;
typedef __attribute__((ext_vector_type(8)))  __bf16       v8bf;
typedef __attribute__((ext_vector_type(16))) __bf16       v16bf;
typedef __attribute__((ext_vector_type(4)))  unsigned int v4u;
typedef __attribute__((ext_vector_type(8)))  unsigned int v8u;

#define NUM_HEAD 16
#define HEAD_DIM 64
#define DIM_IN   1024
#define BATCH    8
#define SEQ      1024
#define HB       (NUM_HEAD * BATCH)          // 128 (h,b) pairs
#define ROWS_QKV (HB * SEQ)                  // 131072 projection rows

__device__ __forceinline__ v8f zero8() {
    v8f z;
#pragma unroll
    for (int i = 0; i < 8; ++i) z[i] = 0.0f;
    return z;
}

// A-matrix fragment, 16x32 bf16 (contiguous K order, matches load_B_32x16):
// lane l: row = l&15, kbase = 16*(l>>4); elems 0..7 -> K=kbase..kbase+7,
// elems 8..15 -> K=kbase+8..kbase+15. Two contiguous 16B loads.
__device__ __forceinline__ v16bf load_A_16x32(const __bf16* base, int ld) {
    const int l   = threadIdx.x & 31;
    const int row = l & 15;
    const int kb  = (l >> 4) * 16;
    const __bf16* p = base + row * ld;
    v8bf lo = *(const v8bf*)(p + kb);
    v8bf hi = *(const v8bf*)(p + kb + 8);
    v16bf a;
#pragma unroll
    for (int i = 0; i < 8; ++i) { a[i] = lo[i]; a[8 + i] = hi[i]; }
    return a;
}

// B-matrix fragment, 32x16 bf16. Source buffer is row-major [n][k]
// (i.e. B[k][n] = buf[n*ld + k]). lane l: col n = l&15, K = 16*(l>>4)+e,
// e = 0..15 -> one contiguous 32B load.
__device__ __forceinline__ v16bf load_B_32x16(const __bf16* base, int ld) {
    const int l   = threadIdx.x & 31;
    const int col = l & 15;
    const int kb  = (l >> 4) * 16;
    return *(const v16bf*)(base + col * ld + kb);
}

__device__ __forceinline__ v8f wmma_bf16(v16bf a, v16bf b, v8f c) {
    return __builtin_amdgcn_wmma_f32_16x16x32_bf16(false, a, false, b,
                                                   (short)0, c, false, false);
}

// Issue a 1-D TDM copy: nbytes (multiple of 4, here 4KB) from global gaddr to
// LDS offset laddr.  D# groups per CDNA5 ISA.
// Must be executed by exactly one wave (TDM ignores EXEC, issues per wave).
__device__ __forceinline__ void tdm_load_1d(unsigned int laddr, unsigned long long gaddr,
                                            unsigned int ndwords) {
    v4u g0;
    g0[0] = 1u;                                   // count=1 (valid user D#)
    g0[1] = laddr;                                // lds_addr (bytes)
    g0[2] = (unsigned int)gaddr;                  // global_addr[31:0]
    g0[3] = (unsigned int)(gaddr >> 32) | (2u << 30);   // addr[56:32] | type=2
    v8u g1;
    g1[0] = (2u << 16);                           // wg_mask=0, data_size=2 (4B)
    g1[1] = (ndwords & 0xffffu) << 16;            // tensor_dim0[15:0]
    g1[2] = (ndwords >> 16) | (1u << 16);         // tensor_dim0[31:16] | tensor_dim1=1
    g1[3] = (ndwords & 0xffffu) << 16;            // tile_dim0 = ndwords
    g1[4] = 0u;                                   // tile_dim1/2 = 0 (1-D)
    g1[5] = ndwords;                              // tensor_dim0_stride
    g1[6] = 0u;
    g1[7] = 0u;
    asm volatile("tensor_load_to_lds %0, %1" :: "s"(g0), "s"(g1) : "memory");
}

// ---------------------------------------------------------------------------
// Kernel 1: shared per-head QKV projection.  Q = xh @ Wq^T + bq (etc.)
// xh[h,b,m,:] = x[b, m, h*64 : h*64+64].  K is pre-scaled by 1/sqrt(1024).
// Output layout: Q/K/V[(h*8+b)*SEQ + m][64] bf16.
// ---------------------------------------------------------------------------
__global__ __launch_bounds__(128)
void mha_qkv_proj_kernel(const float* __restrict__ x,
                         const float* __restrict__ Wq, const float* __restrict__ bq,
                         const float* __restrict__ Wk, const float* __restrict__ bk,
                         const float* __restrict__ Wv, const float* __restrict__ bv,
                         __bf16* __restrict__ Qb, __bf16* __restrict__ Kb,
                         __bf16* __restrict__ Vb) {
    __shared__ __align__(32) __bf16 xs[64 * 64];
    __shared__ __align__(32) __bf16 wqs[64 * 64];
    __shared__ __align__(32) __bf16 wks[64 * 64];
    __shared__ __align__(32) __bf16 wvs[64 * 64];

    const int tid  = threadIdx.x;
    const int wave = tid >> 5;
    const int l    = tid & 31;

    const int Rbase = blockIdx.x * 64;          // 64 rows, same (h,b)
    const int hb    = Rbase >> 10;
    const int m0    = Rbase & (SEQ - 1);
    const int h     = hb >> 3;
    const int b     = hb & 7;

    // Stage x chunk (64 rows x 64 cols) and the three 64x64 weights as bf16.
    for (int e = tid; e < 4096; e += 128) {
        const int r = e >> 6, c = e & 63;
        xs[e]  = (__bf16)x[((size_t)(b * SEQ + m0 + r)) * DIM_IN + h * HEAD_DIM + c];
        wqs[e] = (__bf16)Wq[e];
        wks[e] = (__bf16)Wk[e];
        wvs[e] = (__bf16)Wv[e];
    }
    __syncthreads();

    const __bf16* arow = xs + (wave * 16) * 64;
    const v16bf a0 = load_A_16x32(arow, 64);       // K = 0..31
    const v16bf a1 = load_A_16x32(arow + 32, 64);  // K = 32..63

    const float kscale = 0.03125f;                 // 1/sqrt(1024)
    const int   mg0    = m0 + wave * 16;

#pragma unroll
    for (int nt = 0; nt < 4; ++nt) {
        const int col = nt * 16 + (l & 15);
        // ---- Q ----
        {
            v8f acc = zero8();
            acc = wmma_bf16(a0, load_B_32x16(wqs + nt * 16 * 64, 64), acc);
            acc = wmma_bf16(a1, load_B_32x16(wqs + nt * 16 * 64 + 32, 64), acc);
            const float bias = bq[col];
#pragma unroll
            for (int v = 0; v < 8; ++v) {
                const int M = v + 8 * (l >> 4);
                Qb[(size_t)(hb * SEQ + mg0 + M) * HEAD_DIM + col] = (__bf16)(acc[v] + bias);
            }
        }
        // ---- K (pre-scaled) ----
        {
            v8f acc = zero8();
            acc = wmma_bf16(a0, load_B_32x16(wks + nt * 16 * 64, 64), acc);
            acc = wmma_bf16(a1, load_B_32x16(wks + nt * 16 * 64 + 32, 64), acc);
            const float bias = bk[col];
#pragma unroll
            for (int v = 0; v < 8; ++v) {
                const int M = v + 8 * (l >> 4);
                Kb[(size_t)(hb * SEQ + mg0 + M) * HEAD_DIM + col] =
                    (__bf16)((acc[v] + bias) * kscale);
            }
        }
        // ---- V ----
        {
            v8f acc = zero8();
            acc = wmma_bf16(a0, load_B_32x16(wvs + nt * 16 * 64, 64), acc);
            acc = wmma_bf16(a1, load_B_32x16(wvs + nt * 16 * 64 + 32, 64), acc);
            const float bias = bv[col];
#pragma unroll
            for (int v = 0; v < 8; ++v) {
                const int M = v + 8 * (l >> 4);
                Vb[(size_t)(hb * SEQ + mg0 + M) * HEAD_DIM + col] = (__bf16)(acc[v] + bias);
            }
        }
    }
}

// ---------------------------------------------------------------------------
// Kernel 2: flash attention per (h,b).  scores = Ksc @ Q^T, softmax over n,
// O = A @ V.  Output stored TRANSPOSED: Y[(hb*64 + d)*SEQ + m] bf16, which is
// exactly the reference's transpose(0,1,3,2).reshape(b,m,D) flattening.
// Block: 4 waves, 64 m-rows (16 per wave); loop n in chunks of 32.
// Q tile staged by the Tensor Data Mover (wave 0); V tile staged transposed
// by all threads (TDM cannot transpose 2-byte elements).
// ---------------------------------------------------------------------------
__global__ __launch_bounds__(128)
void mha_attn_kernel(const __bf16* __restrict__ Qb, const __bf16* __restrict__ Kb,
                     const __bf16* __restrict__ Vb, __bf16* __restrict__ Yb) {
    __shared__ __align__(32) __bf16 qs[32 * 64];       // Q rows n..n+31, row-major [n][d]
    __shared__ __align__(32) __bf16 vt[64 * 32];       // V transposed  [d][n]
    __shared__ __align__(32) __bf16 ps[4][16 * 32];    // per-wave P tile [m][n]

    const int tid   = threadIdx.x;
    const int wave  = tid >> 5;
    const int l     = tid & 31;
    const int hb    = blockIdx.y;
    const int mbase = blockIdx.x * 64;

    const __bf16* Kh = Kb + (size_t)hb * SEQ * HEAD_DIM;
    const __bf16* Qh = Qb + (size_t)hb * SEQ * HEAD_DIM;
    const __bf16* Vh = Vb + (size_t)hb * SEQ * HEAD_DIM;

    // Persistent A-fragments: this wave's 16 K-rows (contraction over d=64).
    const __bf16* krow = Kh + (size_t)(mbase + wave * 16) * HEAD_DIM;
    const v16bf ka0 = load_A_16x32(krow, HEAD_DIM);
    const v16bf ka1 = load_A_16x32(krow + 32, HEAD_DIM);

    const unsigned int qs_lds = (unsigned int)(unsigned long long)(void*)qs;

    v8f   o[4];
    float m_run[8], l_run[8];
#pragma unroll
    for (int d = 0; d < 4; ++d) o[d] = zero8();
#pragma unroll
    for (int v = 0; v < 8; ++v) { m_run[v] = -3.0e38f; l_run[v] = 0.0f; }

    for (int nb = 0; nb < SEQ / 32; ++nb) {
        const int n0 = nb * 32;
        __syncthreads();   // previous iteration's LDS reads done
        // Q rows n0..n0+31 = 4KB contiguous -> TDM 1-D descriptor copy.
        if (wave == 0) {
            tdm_load_1d(qs_lds,
                        (unsigned long long)(const void*)(Qh + (size_t)n0 * HEAD_DIM),
                        1024u /* dwords */);
        }
        // V rows staged transposed by all threads; overlaps the TDM copy.
        {
            const __bf16* Vsrc = Vh + (size_t)n0 * HEAD_DIM;
            for (int e = tid; e < 2048; e += 128) {
                const int r = e >> 6, c = e & 63;
                vt[c * 32 + r] = Vsrc[e];
            }
            if (n0 + 32 < SEQ) {   // warm L2 for the next chunk
                __builtin_prefetch(Vh + (size_t)(n0 + 32) * HEAD_DIM + tid * 16, 0, 1);
                __builtin_prefetch(Qh + (size_t)(n0 + 32) * HEAD_DIM + tid * 16, 0, 1);
            }
        }
        if (wave == 0) __builtin_amdgcn_s_wait_tensorcnt(0);
        __syncthreads();

        // S tile: 16 m-rows x 32 n-cols (two 16x16 WMMA tiles, K=64 via 2 calls).
        v8f s[2];
#pragma unroll
        for (int t = 0; t < 2; ++t) {
            v8f acc = zero8();
            acc  = wmma_bf16(ka0, load_B_32x16(qs + t * 16 * 64, 64), acc);
            s[t] = wmma_bf16(ka1, load_B_32x16(qs + t * 16 * 64 + 32, 64), acc);
        }

        // Online softmax (rows live across the 16-lane half-groups).
        float rmax[8];
#pragma unroll
        for (int v = 0; v < 8; ++v) rmax[v] = fmaxf(s[0][v], s[1][v]);
#pragma unroll
        for (int off = 1; off < 16; off <<= 1)
#pragma unroll
            for (int v = 0; v < 8; ++v)
                rmax[v] = fmaxf(rmax[v], __shfl_xor(rmax[v], off, 32));

        __bf16* pw = &ps[wave][0];
        float rsum[8];
#pragma unroll
        for (int v = 0; v < 8; ++v) {
            const float mn   = fmaxf(m_run[v], rmax[v]);
            const float corr = __expf(m_run[v] - mn);
            m_run[v] = mn;
            l_run[v] *= corr;
#pragma unroll
            for (int d = 0; d < 4; ++d) o[d][v] *= corr;
            const float p0 = __expf(s[0][v] - mn);
            const float p1 = __expf(s[1][v] - mn);
            rsum[v] = p0 + p1;
            const int M = v + 8 * (l >> 4);
            pw[M * 32 + (l & 15)]      = (__bf16)p0;
            pw[M * 32 + 16 + (l & 15)] = (__bf16)p1;
        }
#pragma unroll
        for (int off = 1; off < 16; off <<= 1)
#pragma unroll
            for (int v = 0; v < 8; ++v)
                rsum[v] += __shfl_xor(rsum[v], off, 32);
#pragma unroll
        for (int v = 0; v < 8; ++v) l_run[v] += rsum[v];

        // Wave-local LDS RAW: wait for our P stores, then reload in A-layout.
        asm volatile("s_wait_dscnt 0" ::: "memory");
        const v16bf pa = load_A_16x32(pw, 32);   // 16 x 32, K = n-chunk
#pragma unroll
        for (int d = 0; d < 4; ++d)
            o[d] = wmma_bf16(pa, load_B_32x16(vt + d * 16 * 32, 32), o[d]);
    }

    // Normalize and store transposed: Y[(hb*64 + d)][m].
#pragma unroll
    for (int v = 0; v < 8; ++v) {
        const float inv = 1.0f / l_run[v];
        const int   M   = v + 8 * (l >> 4);
        const int   mg  = mbase + wave * 16 + M;
#pragma unroll
        for (int d = 0; d < 4; ++d) {
            const int dc = d * 16 + (l & 15);
            Yb[(size_t)(hb * HEAD_DIM + dc) * SEQ + mg] = (__bf16)(o[d][v] * inv);
        }
    }
}

// ---------------------------------------------------------------------------
// Kernel 3: Wo -> bf16.
// ---------------------------------------------------------------------------
__global__ __launch_bounds__(256)
void mha_cvt_wo_kernel(const float* __restrict__ Wo, __bf16* __restrict__ Wob) {
    const int i = blockIdx.x * 256 + threadIdx.x;
    if (i < DIM_IN * DIM_IN) Wob[i] = (__bf16)Wo[i];
}

// ---------------------------------------------------------------------------
// Kernel 4: out = Y @ Wo^T + bo.  Y is [8192 x 1024] bf16 row-major (the
// scrambled attention output), Wo row-major [o][c].  Operands stream from L2
// (Y 16MB + Wo 2MB resident in 192MB L2).  Block = 64x64 output tile.
// ---------------------------------------------------------------------------
__global__ __launch_bounds__(128)
void mha_out_proj_kernel(const __bf16* __restrict__ Yb, const __bf16* __restrict__ Wob,
                         const float* __restrict__ bo, float* __restrict__ out) {
    const int tid  = threadIdx.x;
    const int wave = tid >> 5;
    const int l    = tid & 31;
    const int rb   = blockIdx.y * 64 + wave * 16;   // 16 rows per wave
    const int cb   = blockIdx.x * 64;

    v8f acc[4];
#pragma unroll
    for (int t = 0; t < 4; ++t) acc[t] = zero8();

    const __bf16* ybase = Yb + (size_t)rb * DIM_IN;
    for (int kk = 0; kk < DIM_IN / 32; ++kk) {
        const v16bf a = load_A_16x32(ybase + kk * 32, DIM_IN);
#pragma unroll
        for (int t = 0; t < 4; ++t) {
            const v16bf bfr =
                load_B_32x16(Wob + (size_t)(cb + t * 16) * DIM_IN + kk * 32, DIM_IN);
            acc[t] = wmma_bf16(a, bfr, acc[t]);
        }
    }

#pragma unroll
    for (int t = 0; t < 4; ++t) {
        const int  col  = cb + t * 16 + (l & 15);
        const float bias = bo[col];
#pragma unroll
        for (int v = 0; v < 8; ++v) {
            const int row = rb + v + 8 * (l >> 4);
            out[(size_t)row * DIM_IN + col] = acc[t][v] + bias;
        }
    }
}

// ---------------------------------------------------------------------------
extern "C" void kernel_launch(void* const* d_in, const int* in_sizes, int n_in,
                              void* d_out, int out_size, void* d_ws, size_t ws_size,
                              hipStream_t stream) {
    (void)in_sizes; (void)n_in; (void)out_size; (void)ws_size;
    const float* x  = (const float*)d_in[0];
    const float* Wq = (const float*)d_in[1];
    const float* bq = (const float*)d_in[2];
    const float* Wk = (const float*)d_in[3];
    const float* bk = (const float*)d_in[4];
    const float* Wv = (const float*)d_in[5];
    const float* bv = (const float*)d_in[6];
    const float* Wo = (const float*)d_in[7];
    const float* bo = (const float*)d_in[8];
    float* out = (float*)d_out;

    char* ws = (char*)d_ws;
    const size_t MB = 1u << 20;
    __bf16* Qb  = (__bf16*)(ws + 0 * MB);    // 16 MB
    __bf16* Kb  = (__bf16*)(ws + 16 * MB);   // 16 MB
    __bf16* Vb  = (__bf16*)(ws + 32 * MB);   // 16 MB
    __bf16* Yb  = (__bf16*)(ws + 48 * MB);   // 16 MB (scrambled attn out)
    __bf16* Wob = (__bf16*)(ws + 64 * MB);   //  2 MB

    mha_qkv_proj_kernel<<<ROWS_QKV / 64, 128, 0, stream>>>(
        x, Wq, bq, Wk, bk, Wv, bv, Qb, Kb, Vb);
    mha_cvt_wo_kernel<<<(DIM_IN * DIM_IN) / 256, 256, 0, stream>>>(Wo, Wob);
    mha_attn_kernel<<<dim3(SEQ / 64, HB), 128, 0, stream>>>(Qb, Kb, Vb, Yb);
    mha_out_proj_kernel<<<dim3(DIM_IN / 64, (HB * HEAD_DIM) / 64), 128, 0, stream>>>(
        Yb, Wob, bo, out);
}
